// AdaptiveGraphLearning_45243185496646
// MI455X (gfx1250) — compile-verified, matrix operand order checked by reading, not run on hardware
//
#include <hip/hip_runtime.h>
#include <hip/hip_bf16.h>
#include <math.h>

#define N_ROWS 8192
#define DIM    128
#define TOPK   11          // K+1 from reference
#define WAVES  4           // waves per block in the simtopk kernel

typedef _Float16     v16h  __attribute__((ext_vector_type(16)));
typedef _Float16     v4h   __attribute__((ext_vector_type(4)));
typedef float        v8f   __attribute__((ext_vector_type(8)));
typedef float        f32x4 __attribute__((ext_vector_type(4)));
typedef unsigned int u32x4 __attribute__((ext_vector_type(4)));

union Frag16 {
    v16h  h;
    u32x4 u[2];
};

// ---------------------------------------------------------------------------
// Kernel 1: L2-normalize rows of X, emit f16 Xn into workspace.
// One wave (32 lanes) per row; 4 floats per lane.
// ---------------------------------------------------------------------------
__global__ void agl_normalize_kernel(const float* __restrict__ X,
                                     _Float16* __restrict__ Xh) {
    const int wave = (blockIdx.x * blockDim.x + threadIdx.x) >> 5;
    const int lane = threadIdx.x & 31;
    if (wave >= N_ROWS) return;

    const f32x4 x = *(const f32x4*)(X + (size_t)wave * DIM + lane * 4);
    float s = x[0] * x[0] + x[1] * x[1] + x[2] * x[2] + x[3] * x[3];
    #pragma unroll
    for (int m = 16; m > 0; m >>= 1) s += __shfl_xor(s, m, 32);

    const float inv = 1.0f / fmaxf(sqrtf(s), 1e-12f);
    v4h h;
    h[0] = (_Float16)(x[0] * inv);
    h[1] = (_Float16)(x[1] * inv);
    h[2] = (_Float16)(x[2] * inv);
    h[3] = (_Float16)(x[3] * inv);
    *(v4h*)(Xh + (size_t)wave * DIM + lane * 4) = h;
}

// ---------------------------------------------------------------------------
// Kernel 2: fused cosine-similarity GEMM (WMMA f16, K=128) + streaming top-11.
// 4 waves per block; each wave owns one quarter (2048 cols = 128 tiles) of the
// 16-row strip, double-buffering B fragments so loads overlap WMMA + top-k.
// Partial top-11 lists are merged once at the end through LDS.
//
// 16-bit A fragment layout (16x32 MxK, wave32):
//   lane l: m = l&15, g = l>>4
//   halves 0..7  hold K = g*8 + 0..7        (16 contiguous bytes)
//   halves 8..15 hold K = g*8 + 16 + 0..7   (16 contiguous bytes)
// B (32x16 KxN) mirrors this with n = l&15; since S = Xn*Xn^T, the B fragment
// for column n is the same loader applied to row n (no transpose needed).
// ---------------------------------------------------------------------------
__global__ void
__launch_bounds__(WAVES * 32, 1)
agl_simtopk_kernel(const _Float16* __restrict__ Xh,
                   float* __restrict__ topv,
                   int*   __restrict__ topi) {
    __shared__ float tile[WAVES][16][20];      // 20-float stride: 16B-aligned rows
    __shared__ float pv[WAVES][16][TOPK];
    __shared__ int   pi_s[WAVES][16][TOPK];

    const int r0     = blockIdx.x * 16;
    const int waveId = threadIdx.x >> 5;
    const int lane   = threadIdx.x & 31;
    const int g      = lane >> 4;
    const int mr     = lane & 15;

    // A fragments for this 16-row strip (4 K-chunks of 32) -- shared by design,
    // each wave keeps its own register copy.
    Frag16 afrag[4];
    {
        const _Float16* arow = Xh + (size_t)(r0 + mr) * DIM;
        #pragma unroll
        for (int c = 0; c < 4; ++c) {
            afrag[c].u[0] = *(const u32x4*)(arow + c * 32 + g * 8);
            afrag[c].u[1] = *(const u32x4*)(arow + c * 32 + g * 8 + 16);
        }
    }

    // Per-thread top-11 (descending), registers only (static indexing).
    float tv[TOPK];
    int   ti[TOPK];
    #pragma unroll
    for (int k = 0; k < TOPK; ++k) { tv[k] = -2.0f; ti[k] = -1; }

    auto insert11 = [&](float v, int col) {
        if (v > tv[TOPK - 1]) {
            #pragma unroll
            for (int k = 0; k < TOPK; ++k) {
                const bool gt = v > tv[k];
                const float nv = gt ? v   : tv[k];
                const int   ni = gt ? col : ti[k];
                v     = gt ? tv[k] : v;
                col   = gt ? ti[k] : col;
                tv[k] = nv;
                ti[k] = ni;
            }
        }
    };

    const int TILES = N_ROWS / (16 * WAVES);   // 128 column tiles per wave
    const int tbase = waveId * TILES;          // this wave's first tile index

    auto loadB = [&](int t, Frag16* f) {
        const _Float16* brow = Xh + (size_t)((tbase + t) * 16 + mr) * DIM;
        #pragma unroll
        for (int c = 0; c < 4; ++c) {
            f[c].u[0] = *(const u32x4*)(brow + c * 32 + g * 8);
            f[c].u[1] = *(const u32x4*)(brow + c * 32 + g * 8 + 16);
        }
    };

    auto compute = [&](const Frag16* f, int t) {
        v8f acc = {};
        #pragma unroll
        for (int c = 0; c < 4; ++c)
            acc = __builtin_amdgcn_wmma_f32_16x16x32_f16(
                false, afrag[c].h, false, f[c].h,
                (short)0, acc, false, false);

        // D layout: VGPR i -> row (g*8 + i), col = lane&15.  Per-wave tile,
        // same-wave DS ops are in-order: a dscnt wait (not a barrier) suffices.
        #pragma unroll
        for (int i = 0; i < 8; ++i)
            tile[waveId][g * 8 + i][mr] = acc[i];
        asm volatile("s_wait_dscnt 0x0" ::: "memory");

        if (lane < 16) {
            const f32x4* rp = (const f32x4*)&tile[waveId][lane][0];
            const f32x4 q0 = rp[0], q1 = rp[1], q2 = rp[2], q3 = rp[3];
            const int c0 = (tbase + t) * 16;
            insert11(q0[0], c0 + 0);  insert11(q0[1], c0 + 1);
            insert11(q0[2], c0 + 2);  insert11(q0[3], c0 + 3);
            insert11(q1[0], c0 + 4);  insert11(q1[1], c0 + 5);
            insert11(q1[2], c0 + 6);  insert11(q1[3], c0 + 7);
            insert11(q2[0], c0 + 8);  insert11(q2[1], c0 + 9);
            insert11(q2[2], c0 + 10); insert11(q2[3], c0 + 11);
            insert11(q3[0], c0 + 12); insert11(q3[1], c0 + 13);
            insert11(q3[2], c0 + 14); insert11(q3[3], c0 + 15);
        }
        asm volatile("" ::: "memory");   // keep next tile's DS stores after reads
    };

    // Ping-pong double buffering: next tile's 8 b128 loads issue while the
    // current tile's WMMAs + top-k run.
    Frag16 bf[4], bn[4];
    loadB(0, bf);
    for (int t = 0; t < TILES; t += 2) {
        loadB(t + 1, bn);                     // TILES is even: t+1 < TILES
        compute(bf, t);
        if (t + 2 < TILES) loadB(t + 2, bf);
        compute(bn, t + 1);
    }

    // ---- merge the 4 per-wave partial lists --------------------------------
    if (lane < 16) {
        #pragma unroll
        for (int k = 0; k < TOPK; ++k) {
            pv[waveId][lane][k]   = tv[k];
            pi_s[waveId][lane][k] = ti[k];
        }
    }
    __syncthreads();

    if (threadIdx.x < 16) {
        const int t   = threadIdx.x;
        const int row = r0 + t;
        float mv[TOPK]; int mi[TOPK];
        #pragma unroll
        for (int k = 0; k < TOPK; ++k) { mv[k] = pv[0][t][k]; mi[k] = pi_s[0][t][k]; }

        #pragma unroll
        for (int w = 1; w < WAVES; ++w) {
            #pragma unroll
            for (int k = 0; k < TOPK; ++k) {
                float v  = pv[w][t][k];
                int  col = pi_s[w][t][k];
                if (v > mv[TOPK - 1]) {
                    #pragma unroll
                    for (int q = 0; q < TOPK; ++q) {
                        const bool gt = v > mv[q];
                        const float nv = gt ? v   : mv[q];
                        const int   ni = gt ? col : mi[q];
                        v     = gt ? mv[q] : v;
                        col   = gt ? mi[q] : col;
                        mv[q] = nv;
                        mi[q] = ni;
                    }
                }
            }
        }

        float sum = 1e-6f;
        #pragma unroll
        for (int k = 0; k < TOPK; ++k) {
            if (mi[k] == row) mv[k] = 0.0f;   // drop self-edge (value -> 0)
            sum += mv[k];
        }
        const float inv = 1.0f / sum;
        #pragma unroll
        for (int k = 0; k < TOPK; ++k) {
            topv[row * TOPK + k] = mv[k] * inv;
            topi[row * TOPK + k] = mi[k];
        }
    }
}

// ---------------------------------------------------------------------------
// Kernel 3a: the bandwidth-critical pass (~800 MB -> ~35 us at 23.3 TB/s).
// A_final = sigmoid(lam)*A_raw ; A_learned = 0.  Non-temporal f32x4 streams
// so the write-once output doesn't thrash L2.
// ---------------------------------------------------------------------------
__global__ void agl_blend_base_kernel(const float* __restrict__ A_raw,
                                      const float* __restrict__ lam_p,
                                      float* __restrict__ out) {
    const float lam = 1.0f / (1.0f + expf(-lam_p[0]));
    const size_t i4 = (size_t)blockIdx.x * blockDim.x + threadIdx.x;
    const size_t total = (size_t)N_ROWS * N_ROWS;

    const f32x4 a = __builtin_nontemporal_load((const f32x4*)A_raw + i4);
    f32x4 f;
    f[0] = lam * a[0]; f[1] = lam * a[1];
    f[2] = lam * a[2]; f[3] = lam * a[3];
    __builtin_nontemporal_store(f, (f32x4*)out + i4);

    const f32x4 z = {0.f, 0.f, 0.f, 0.f};
    __builtin_nontemporal_store(z, (f32x4*)(out + total) + i4);
}

// ---------------------------------------------------------------------------
// Kernel 3b: scatter the 11 sparse entries per row into both outputs.
// ---------------------------------------------------------------------------
__global__ void agl_scatter_kernel(const float* __restrict__ topv,
                                   const int*   __restrict__ topi,
                                   const float* __restrict__ A_raw,
                                   const float* __restrict__ lam_p,
                                   float* __restrict__ out) {
    const int t = blockIdx.x * blockDim.x + threadIdx.x;
    if (t >= N_ROWS * TOPK) return;
    const int row = t / TOPK;
    const int col = topi[t];
    if (col < 0 || col == row) return;

    const float v   = topv[t];
    const float lam = 1.0f / (1.0f + expf(-lam_p[0]));
    const size_t off   = (size_t)row * N_ROWS + col;
    const size_t total = (size_t)N_ROWS * N_ROWS;

    out[off]         = lam * A_raw[off] + (1.0f - lam) * v;  // A_final
    out[total + off] = v;                                    // A_learned
}

// ---------------------------------------------------------------------------
extern "C" void kernel_launch(void* const* d_in, const int* in_sizes, int n_in,
                              void* d_out, int out_size, void* d_ws, size_t ws_size,
                              hipStream_t stream) {
    const float* X     = (const float*)d_in[0];
    const float* A_raw = (const float*)d_in[1];
    const float* lam_p = (const float*)d_in[2];
    float* out = (float*)d_out;

    // Workspace layout (offsets 256B-aligned):
    //   [0, 2MB)       : Xn in f16   (8192*128*2)
    //   [2MB, +352KB)  : topv f32    (8192*11*4)
    //   [next, +352KB) : topi i32    (8192*11*4)
    char* ws = (char*)d_ws;
    _Float16* Xh = (_Float16*)ws;
    size_t off = (size_t)N_ROWS * DIM * sizeof(_Float16);
    float* topv = (float*)(ws + off);
    off += ((size_t)N_ROWS * TOPK * sizeof(float) + 255) & ~(size_t)255;
    int* topi = (int*)(ws + off);

    // 1) normalize: one wave per row -> 8192 waves
    agl_normalize_kernel<<<N_ROWS / 8, 256, 0, stream>>>(X, Xh);

    // 2) WMMA similarity + streaming top-11: 4 waves per 16-row strip
    agl_simtopk_kernel<<<N_ROWS / 16, WAVES * 32, 0, stream>>>(Xh, topv, topi);

    // 3a) non-temporal streaming blend/zero pass: N*N/4 f32x4 groups
    const size_t groups = ((size_t)N_ROWS * N_ROWS) / 4;   // 16,777,216
    agl_blend_base_kernel<<<(unsigned)(groups / 256), 256, 0, stream>>>(A_raw, lam_p, out);

    // 3b) scatter sparse top-k entries
    const int nscatter = N_ROWS * TOPK;
    agl_scatter_kernel<<<(nscatter + 255) / 256, 256, 0, stream>>>(topv, topi, A_raw, lam_p, out);
}
